// EdgeConvHeadRefactorFusion_21406117004181
// MI455X (gfx1250) — matrix operand order repaired and checked by reading, not compiled, not run on hardware
//
#include <hip/hip_runtime.h>

// ---------------------------------------------------------------------------
// EdgeConv head for MI455X (gfx1250): all GEMMs on v_wmma_f32_16x16x32_bf16.
// A packed-bf16 feature cache (hfB) removes all float->bf16 conversion work
// from the N^2 kNN loop; remaining packs use single v_perm_b32 (RTZ bf16).
// ---------------------------------------------------------------------------

typedef __attribute__((ext_vector_type(16))) __bf16 v16bf;
typedef __attribute__((ext_vector_type(8)))  float  v8f;

static constexpr int BATCH = 2;
static constexpr int NPTS  = 8192;
static constexpr int DCH   = 64;
static constexpr int KNN   = 20;
static constexpr float EPS = 1e-5f;

// bf16 truncation (RTZ): top 16 bits. Single shift / v_perm; precision is
// bf16-limited anyway and kNN ordering is insensitive to RTZ-vs-RNE.
static __device__ __forceinline__ unsigned short bf16_trunc(float f) {
    return (unsigned short)(__builtin_bit_cast(unsigned, f) >> 16);
}
// Pack two floats to packed bf16x2 in ONE v_perm_b32:
// dst = { b[31:16], a[31:16] }  (a in low half)
static __device__ __forceinline__ unsigned pack2bf(float a, float b) {
    return __builtin_amdgcn_perm(__builtin_bit_cast(unsigned, b),
                                 __builtin_bit_cast(unsigned, a),
                                 0x07060302u);
}

static __device__ __forceinline__ v8f wmma_bf16(v16bf a, v16bf b, v8f c) {
    return __builtin_amdgcn_wmma_f32_16x16x32_bf16(false, a, false, b,
                                                   (short)0, c, false, false);
}

union FragU {
    v16bf v;
    uint2 q[4];
    unsigned u[8];
};

// A fragment (16 x 32 bf16) from LDS tile[row][c] (2 bf16/dword), row stride
// strideDW (even -> 8B-aligned b64 loads).
//   lane: m = lane&15, h = lane>>4
//   u[0..7] = dwords ks*16 + 8h + {0..7}
static __device__ __forceinline__ v16bf load_afrag_lds(const unsigned* tile,
                                                       int row0, int ks,
                                                       int lane, int strideDW) {
    const int m = lane & 15, h = lane >> 4;
    const unsigned* base = tile + (row0 + m) * strideDW + ks * 16 + 8 * h;
    FragU f;
    f.q[0] = *(const uint2*)(base + 0);
    f.q[1] = *(const uint2*)(base + 2);
    f.q[2] = *(const uint2*)(base + 4);
    f.q[3] = *(const uint2*)(base + 6);
    return f.v;
}

// B fragment (32 x 16 bf16) from LDS tile[col][c]:
//   u[0..7] = dwords ks*16 + 8h + {0..7}  (8B-aligned b64 loads)
static __device__ __forceinline__ v16bf load_bfrag_lds(const unsigned* tile,
                                                       int col0, int ks,
                                                       int lane, int strideDW) {
    const int n = lane & 15, h = lane >> 4;
    const unsigned* base = tile + (col0 + n) * strideDW + ks * 16 + 8 * h;
    FragU f;
    f.q[0] = *(const uint2*)(base + 0);
    f.q[1] = *(const uint2*)(base + 2);
    f.q[2] = *(const uint2*)(base + 4);
    f.q[3] = *(const uint2*)(base + 6);
    return f.v;
}

// ---------------------------------------------------------------------------
// Kernel 1: brute-force kNN on the C=3 input cloud (VALU; K too small for WMMA)
// ---------------------------------------------------------------------------
__global__ void __launch_bounds__(128)
knn3_kernel(const float* __restrict__ x, int* __restrict__ idxb) {
    __shared__ float col[256 * 4];

    const int t = threadIdx.x;
    const int b = blockIdx.x / (NPTS / 128);
    const int i = (blockIdx.x % (NPTS / 128)) * 128 + t;

    const float cx = x[(size_t)(b * 3 + 0) * NPTS + i];
    const float cy = x[(size_t)(b * 3 + 1) * NPTS + i];
    const float cz = x[(size_t)(b * 3 + 2) * NPTS + i];

    float bv[KNN];
    int   bi[KNN];
#pragma unroll
    for (int k = 0; k < KNN; ++k) { bv[k] = -1e30f; bi[k] = 0; }

    for (int j0 = 0; j0 < NPTS; j0 += 256) {
        __syncthreads();
#pragma unroll
        for (int q = 0; q < 6; ++q) {          // stage 3 x 256 coords
            const int flat = q * 128 + t;
            const int c = flat >> 8, n = flat & 255;
            col[n * 4 + c] = x[(size_t)(b * 3 + c) * NPTS + j0 + n];
        }
        __syncthreads();
        for (int n = 0; n < 256; ++n) {
            const float dx = cx - col[n * 4 + 0];
            const float dy = cy - col[n * 4 + 1];
            const float dz = cz - col[n * 4 + 2];
            const float v  = -(dx * dx + dy * dy + dz * dz);
            if (v > bv[0]) {
                bv[0] = v; bi[0] = j0 + n;
#pragma unroll
                for (int s = 0; s < KNN - 1; ++s) {
                    if (bv[s] > bv[s + 1]) {
                        float tv = bv[s]; bv[s] = bv[s + 1]; bv[s + 1] = tv;
                        int   ti = bi[s]; bi[s] = bi[s + 1]; bi[s + 1] = ti;
                    }
                }
            }
        }
    }
    int* op = idxb + ((size_t)b * NPTS + i) * KNN;
#pragma unroll
    for (int k = 0; k < KNN; ++k) op[k] = bi[KNN - 1 - k];
}

// ---------------------------------------------------------------------------
// Kernel 2: first edge-conv (2C=6 -> 64), tiny GEMM per point; VALU.
// ---------------------------------------------------------------------------
__global__ void __launch_bounds__(64)
edgeconv0_kernel(const float* __restrict__ x, const int* __restrict__ idxb,
                 const float* __restrict__ W, const float* __restrict__ g,
                 const float* __restrict__ be, const float* __restrict__ mu,
                 const float* __restrict__ va, float* __restrict__ out0) {
    __shared__ float ef[KNN][8];   // 6 features per edge (padded)

    const int t = threadIdx.x;
    const int b = blockIdx.x / NPTS;
    const int n = blockIdx.x % NPTS;

    if (t < KNN) {
        const int j = idxb[((size_t)b * NPTS + n) * KNN + t];
#pragma unroll
        for (int c = 0; c < 3; ++c) {
            const float ctr = x[(size_t)(b * 3 + c) * NPTS + n];
            const float nb  = x[(size_t)(b * 3 + c) * NPTS + j];
            ef[t][c]     = nb - ctr;
            ef[t][3 + c] = ctr;
        }
    }
    __syncthreads();

    const int o = t;
    const float* w = W + o * 6;
    float mx = -1e30f;
#pragma unroll 4
    for (int k = 0; k < KNN; ++k) {
        const float d = ef[k][0] * w[0] + ef[k][1] * w[1] + ef[k][2] * w[2] +
                        ef[k][3] * w[3] + ef[k][4] * w[4] + ef[k][5] * w[5];
        mx = fmaxf(mx, d);
    }
    const float s = g[o] * rsqrtf(va[o] + EPS);
    float v = (mx - mu[o]) * s + be[o];
    v = (v >= 0.0f) ? v : 0.2f * v;
    out0[((size_t)b * DCH + o) * NPTS + n] = v;
}

// ---------------------------------------------------------------------------
// Kernel 3: fuse layer  hf = elu(fuse_w @ cat(h, reps))  via bf16 WMMA.
// One wave = one 16(o) x 16(n) tile; two independent accumulator chains
// break the WMMA->WMMA hazard. Writes BOTH f32 hfT[b][n][c] and the packed
// bf16 cache hfB[b][n][c/2] (a lane holds 8 consecutive channels -> 4 dwords).
// ---------------------------------------------------------------------------
__global__ void __launch_bounds__(32)
fuse_kernel(const float* __restrict__ hprev,  // (B,64,N) = previous out slice
            const float* __restrict__ reps,   // (B,64,N) for this iteration
            const float* __restrict__ W,      // (64,128)
            float* __restrict__ hfT,          // (B,N,64) f32
            unsigned* __restrict__ hfB) {     // (B,N,32) packed bf16 pairs
    const int lane  = threadIdx.x;
    const int id    = blockIdx.x;                 // 0..4095
    const int ntile = id & 511;                   // N/16
    const int ot    = (id >> 9) & 3;
    const int b     = id >> 11;
    const int n0    = ntile * 16;
    const int o0    = ot * 16;
    const int ln    = lane & 15, h = lane >> 4;

    v8f acc0 = {0.f, 0.f, 0.f, 0.f, 0.f, 0.f, 0.f, 0.f};
    v8f acc1 = {0.f, 0.f, 0.f, 0.f, 0.f, 0.f, 0.f, 0.f};
    const float* wrow = W + (o0 + ln) * 128;

#pragma unroll
    for (int ks = 0; ks < 4; ++ks) {
        // base pointer + channel offset are compile-time uniform per ks
        const float* bsrc = (ks < 2) ? hprev : reps;
        const int    cb   = (ks & 1) * 32;
        FragU a, bb;
#pragma unroll
        for (int j = 0; j < 8; ++j) {            // A = weights (o x c)
            const int kk = 16 * h + 2 * j;
            const int c  = ks * 32 + kk;
            const float2 p = *(const float2*)(wrow + c);
            a.u[j] = pack2bf(p.x, p.y);
        }
#pragma unroll
        for (int j = 0; j < 8; ++j) {            // B = cat(h, reps) (c x n)
            const int c = cb + 16 * h + 2 * j;
            const float* r0 = bsrc + ((size_t)b * DCH + c) * NPTS;
            bb.u[j] = pack2bf(r0[n0 + ln], r0[NPTS + n0 + ln]);  // c and c+1
        }
        if (ks & 1) acc1 = wmma_bf16(a.v, bb.v, acc1);
        else        acc0 = wmma_bf16(a.v, bb.v, acc0);
    }

    const int n = n0 + ln;
    float elu[8];
#pragma unroll
    for (int r = 0; r < 8; ++r) {
        const float v = acc0[r] + acc1[r];
        elu[r] = (v > 0.0f) ? v : expm1f(v);     // ELU(alpha=1)
    }
    float* dst = hfT + ((size_t)b * NPTS + n) * DCH + o0 + 8 * h;
#pragma unroll
    for (int r = 0; r < 8; ++r) dst[r] = elu[r]; // channels o0+8h .. +7

    unsigned* bq = hfB + ((size_t)b * NPTS + n) * 32 + (o0 >> 1) + 4 * h;
#pragma unroll
    for (int q = 0; q < 4; ++q)
        bq[q] = pack2bf(elu[2 * q], elu[2 * q + 1]);
}

// ---------------------------------------------------------------------------
// Kernel 4: per-point squared norms of hfT rows.
// ---------------------------------------------------------------------------
__global__ void __launch_bounds__(256)
sqnorm_kernel(const float* __restrict__ hfT, float* __restrict__ sqn) {
    const int t = blockIdx.x * 256 + threadIdx.x;   // 0 .. B*N-1
    const float4* r = (const float4*)(hfT + (size_t)t * DCH);
    float s = 0.0f;
#pragma unroll
    for (int q = 0; q < 16; ++q) {
        const float4 p = r[q];
        s += p.x * p.x + p.y * p.y + p.z * p.z + p.w * p.w;
    }
    sqn[t] = s;
}

// ---------------------------------------------------------------------------
// Kernel 5: kNN on 64-channel features. One wave owns 32 query rows.
// LDS tiles staged by pure dword copies from the bf16 cache (no conversions
// in the N^2 loop). Distance tiles via WMMA; streaming top-20 per row.
// ---------------------------------------------------------------------------
__global__ void __launch_bounds__(32)
knn64_kernel(const unsigned* __restrict__ hfB, const float* __restrict__ sqn,
             int* __restrict__ idxb) {
    __shared__ unsigned rowT[32 * 34];   // bf16 [row][c], 2/dword, stride 34
    __shared__ unsigned colT[64 * 34];   // bf16 [col][c]
    __shared__ float    dist[32 * 66];
    __shared__ float    sqr[32];
    __shared__ float    sqc[64];

    const int lane = threadIdx.x;
    const int blk  = blockIdx.x;                  // 0 .. B*N/32-1
    const int b    = blk / (NPTS / 32);
    const int i0   = (blk % (NPTS / 32)) * 32;

    {   // stage this block's 32 query rows (pure copy, 128B each)
        const uint4* src = (const uint4*)(hfB + ((size_t)b * NPTS + i0 + lane) * 32);
        unsigned* dr = &rowT[lane * 34];
#pragma unroll
        for (int q = 0; q < 8; ++q) {
            const uint4 p = src[q];
            *(uint2*)(dr + 4 * q)     = make_uint2(p.x, p.y);
            *(uint2*)(dr + 4 * q + 2) = make_uint2(p.z, p.w);
        }
        sqr[lane] = sqn[b * NPTS + i0 + lane];
    }
    __syncthreads();

    // Preload loop-invariant A fragments (2 M-tiles x 2 K-steps)
    v16bf afr[2][2];
#pragma unroll
    for (int mt = 0; mt < 2; ++mt)
#pragma unroll
        for (int ks = 0; ks < 2; ++ks)
            afr[mt][ks] = load_afrag_lds(rowT, mt * 16, ks, lane, 34);

    // Loop-invariant query norms for this lane's C-fragment rows
    const int hv = (lane >> 4) * 8;
    const int nl = lane & 15;
    float srA[8], srB[8];
#pragma unroll
    for (int r = 0; r < 8; ++r) {
        srA[r] = sqr[r + hv];
        srB[r] = sqr[16 + r + hv];
    }

    float bv[KNN];
    int   bi[KNN];
#pragma unroll
    for (int k = 0; k < KNN; ++k) { bv[k] = -1e30f; bi[k] = 0; }

    for (int j0 = 0; j0 < NPTS; j0 += 64) {
#pragma unroll
        for (int rr = 0; rr < 2; ++rr) {          // stage 64 candidate columns
            const int n = lane + 32 * rr;
            const uint4* src = (const uint4*)(hfB + ((size_t)b * NPTS + j0 + n) * 32);
            unsigned* dc = &colT[n * 34];
#pragma unroll
            for (int q = 0; q < 8; ++q) {
                const uint4 p = src[q];
                *(uint2*)(dc + 4 * q)     = make_uint2(p.x, p.y);
                *(uint2*)(dc + 4 * q + 2) = make_uint2(p.z, p.w);
            }
            sqc[n] = sqn[b * NPTS + j0 + n];
        }
        if (j0 + 64 < NPTS) {                     // prefetch next candidate chunk
            __builtin_prefetch(hfB + ((size_t)b * NPTS + j0 + 64 + lane) * 32, 0, 1);
            __builtin_prefetch(hfB + ((size_t)b * NPTS + j0 + 96 + lane) * 32, 0, 1);
        }
        __syncthreads();

#pragma unroll
        for (int nt = 0; nt < 4; ++nt) {
            v8f acc0 = {0.f, 0.f, 0.f, 0.f, 0.f, 0.f, 0.f, 0.f};
            v8f acc1 = {0.f, 0.f, 0.f, 0.f, 0.f, 0.f, 0.f, 0.f};
#pragma unroll
            for (int ks = 0; ks < 2; ++ks) {
                const v16bf bf = load_bfrag_lds(colT, nt * 16, ks, lane, 34);
                acc0 = wmma_bf16(afr[0][ks], bf, acc0);
                acc1 = wmma_bf16(afr[1][ks], bf, acc1);
            }
            const int n_loc = nt * 16 + nl;
            const float sc  = sqc[n_loc];
#pragma unroll
            for (int r = 0; r < 8; ++r) {
                dist[(r + hv) * 66 + n_loc]      = 2.0f * acc0[r] - srA[r] - sc;
                dist[(16 + r + hv) * 66 + n_loc] = 2.0f * acc1[r] - srB[r] - sc;
            }
        }
        __syncthreads();

        // streaming top-20 (row = lane), 2 candidates per LDS b64 read
        const float2* drow = (const float2*)&dist[lane * 66];
        for (int n2 = 0; n2 < 32; ++n2) {
            const float2 v2 = drow[n2];
#pragma unroll
            for (int half = 0; half < 2; ++half) {
                const float v = half ? v2.y : v2.x;
                if (v > bv[0]) {
                    bv[0] = v; bi[0] = j0 + 2 * n2 + half;
#pragma unroll
                    for (int s = 0; s < KNN - 1; ++s) {
                        if (bv[s] > bv[s + 1]) {
                            float tv = bv[s]; bv[s] = bv[s + 1]; bv[s + 1] = tv;
                            int   ti = bi[s]; bi[s] = bi[s + 1]; bi[s + 1] = ti;
                        }
                    }
                }
            }
        }
        __syncthreads();
    }

    int* op = idxb + ((size_t)b * NPTS + i0 + lane) * KNN;
#pragma unroll
    for (int k = 0; k < KNN; ++k) op[k] = bi[KNN - 1 - k];
}

// ---------------------------------------------------------------------------
// Kernel 6: edge-conv (2C=128 -> 64). One wave = 4 points = 80 edges
// = exactly 5 dense M-tiles; 5x4x4 = 80 WMMAs per wave.
// ---------------------------------------------------------------------------
__global__ void __launch_bounds__(32)
edgeconv64_kernel(const float* __restrict__ hfT, const int* __restrict__ idxb,
                  const float* __restrict__ W,   // (64,128)
                  const float* __restrict__ g, const float* __restrict__ be,
                  const float* __restrict__ mu, const float* __restrict__ va,
                  float* __restrict__ outslice) {
    __shared__ unsigned featT[80 * 66];   // bf16 [edge][c], 128 ch, stride 66 DW
    __shared__ float    eo[80 * 66];      // raw conv output [edge][o]
    __shared__ int      nid[80];

    const int lane = threadIdx.x;
    const int blk  = blockIdx.x;                  // 0 .. B*N/4-1
    const int b    = blk / (NPTS / 4);
    const int base = (blk % (NPTS / 4)) * 4;

    for (int e = lane; e < 80; e += 32)
        nid[e] = idxb[((size_t)b * NPTS + base + e / KNN) * KNN + (e % KNN)];
    __syncthreads();

    // Stage edge features: [nbr-ctr (64) | ctr (64)] as bf16, transposed.
    const int c0 = lane, c1 = lane + 32;
#pragma unroll
    for (int p = 0; p < 4; ++p) {
        const float* ctr = hfT + ((size_t)b * NPTS + base + p) * DCH;
        const float cv0 = ctr[c0], cv1 = ctr[c1];
        const unsigned short cb0 = bf16_trunc(cv0);
        const unsigned short cb1 = bf16_trunc(cv1);
        for (int k = 0; k < KNN; ++k) {
            const int e = p * KNN + k;
            const float* nb = hfT + ((size_t)b * NPTS + nid[e]) * DCH;
            unsigned short* fr = (unsigned short*)&featT[e * 66];
            fr[c0]      = bf16_trunc(nb[c0] - cv0);
            fr[c1]      = bf16_trunc(nb[c1] - cv1);
            fr[64 + c0] = cb0;
            fr[64 + c1] = cb1;
        }
    }
    __syncthreads();

    const int ln = lane & 15, h = lane >> 4;

#pragma unroll
    for (int nt = 0; nt < 4; ++nt) {              // output-channel tiles
        v8f acc[5];
#pragma unroll
        for (int mt = 0; mt < 5; ++mt)
            acc[mt] = (v8f){0.f, 0.f, 0.f, 0.f, 0.f, 0.f, 0.f, 0.f};

        const float* wrow = W + (nt * 16 + ln) * 128;   // B[c][o] = W[o][c]
#pragma unroll
        for (int ks = 0; ks < 4; ++ks) {
            FragU bf;
#pragma unroll
            for (int j = 0; j < 8; ++j) {
                const int c = ks * 32 + 16 * h + 2 * j;
                const float2 p = *(const float2*)(wrow + c);
                bf.u[j] = pack2bf(p.x, p.y);
            }
#pragma unroll
            for (int mt = 0; mt < 5; ++mt) {
                const v16bf af = load_afrag_lds(featT, mt * 16, ks, lane, 66);
                acc[mt] = wmma_bf16(af, bf.v, acc[mt]);
            }
        }
#pragma unroll
        for (int mt = 0; mt < 5; ++mt) {
            const int ocol = nt * 16 + ln;
#pragma unroll
            for (int r = 0; r < 8; ++r) {
                const int e = mt * 16 + r + 8 * h;
                eo[e * 66 + ocol] = acc[mt][r];
            }
        }
    }
    __syncthreads();

    // Max over 20 edges, then BN + LeakyReLU (monotone, gamma > 0)
#pragma unroll
    for (int q = 0; q < 8; ++q) {
        const int flat = lane + 32 * q;           // 4 points x 64 channels
        const int p = flat >> 6, o = flat & 63;
        float mx = -1e30f;
#pragma unroll 4
        for (int k = 0; k < KNN; ++k)
            mx = fmaxf(mx, eo[(p * KNN + k) * 66 + o]);
        const float s = g[o] * rsqrtf(va[o] + EPS);
        float v = (mx - mu[o]) * s + be[o];
        v = (v >= 0.0f) ? v : 0.2f * v;
        outslice[((size_t)b * DCH + o) * NPTS + base + p] = v;
    }
}

// ---------------------------------------------------------------------------
// Host-side launcher
// ---------------------------------------------------------------------------
extern "C" void kernel_launch(void* const* d_in, const int* in_sizes, int n_in,
                              void* d_out, int out_size, void* d_ws, size_t ws_size,
                              hipStream_t stream) {
    (void)in_sizes; (void)n_in; (void)out_size; (void)ws_size;

    const float* x       = (const float*)d_in[0];   // (B,3,N)
    const float* reps    = (const float*)d_in[1];   // (3,B,64,N)
    const float* conv_w0 = (const float*)d_in[2];   // (64,6)
    const float* conv_ws = (const float*)d_in[3];   // (3,64,128)
    const float* gammas  = (const float*)d_in[4];   // (4,64)
    const float* betas   = (const float*)d_in[5];
    const float* run_m   = (const float*)d_in[6];
    const float* run_v   = (const float*)d_in[7];
    const float* fuse_ws = (const float*)d_in[8];   // (3,64,128)
    float* out = (float*)d_out;                     // 4 x (B,64,N)

    char* ws = (char*)d_ws;
    size_t off = 0;
    float* hfT = (float*)(ws + off);  off += sizeof(float) * (size_t)BATCH * NPTS * DCH;
    float* sqn = (float*)(ws + off);  off += sizeof(float) * (size_t)BATCH * NPTS;
    int*   idxb = (int*)(ws + off);   off += sizeof(int) * (size_t)BATCH * NPTS * KNN;
    unsigned* hfB = (unsigned*)(ws + off);  // B*N*32 dwords (packed bf16)

    const size_t sliceSz = (size_t)BATCH * DCH * NPTS;

    knn3_kernel<<<BATCH * (NPTS / 128), 128, 0, stream>>>(x, idxb);
    edgeconv0_kernel<<<BATCH * NPTS, 64, 0, stream>>>(
        x, idxb, conv_w0, gammas, betas, run_m, run_v, out);

    for (int i = 0; i < 3; ++i) {
        const float* hprev = out + (size_t)i * sliceSz;
        fuse_kernel<<<BATCH * 4 * (NPTS / 16), 32, 0, stream>>>(
            hprev, reps + (size_t)i * BATCH * DCH * NPTS,
            fuse_ws + (size_t)i * DCH * 2 * DCH, hfT, hfB);
        sqnorm_kernel<<<(BATCH * NPTS) / 256, 256, 0, stream>>>(hfT, sqn);
        knn64_kernel<<<BATCH * (NPTS / 32), 32, 0, stream>>>(hfB, sqn, idxb);
        edgeconv64_kernel<<<BATCH * (NPTS / 4), 32, 0, stream>>>(
            hfT, idxb, conv_ws + (size_t)i * DCH * 2 * DCH,
            gammas + (size_t)(i + 1) * DCH, betas + (size_t)(i + 1) * DCH,
            run_m + (size_t)(i + 1) * DCH, run_v + (size_t)(i + 1) * DCH,
            out + (size_t)(i + 1) * sliceSz);
    }
}
